// MultiheadAttention_29489245454970
// MI455X (gfx1250) — matrix operand order repaired and checked by reading, not compiled, not run on hardware
//
#include <hip/hip_runtime.h>
#include <hip/hip_bf16.h>
#include <math.h>

typedef __attribute__((ext_vector_type(16))) _Float16 v16h;
typedef __attribute__((ext_vector_type(8)))  _Float16 v8h;
typedef __attribute__((ext_vector_type(8)))  float    v8f;
typedef __attribute__((ext_vector_type(4)))  int      v4i;

#define D_MODEL 1024
#define NHEAD   16
#define BATCH   4
#define SEQ     2048
#define DHEAD   64

// ---------------------------------------------------------------------------
// gfx1250 async global->LDS copy path (ASYNCcnt), with portable fallback.
#if defined(__has_builtin)
#if __has_builtin(__builtin_amdgcn_global_load_async_to_lds_b128)
#define USE_ASYNC_COPY 1
#endif
#if __has_builtin(__builtin_amdgcn_s_wait_asynccnt)
#define HAVE_WAIT_ASYNC_BUILTIN 1
#endif
#endif
#ifndef USE_ASYNC_COPY
#define USE_ASYNC_COPY 0
#endif
#ifndef HAVE_WAIT_ASYNC_BUILTIN
#define HAVE_WAIT_ASYNC_BUILTIN 0
#endif

#define AS3 __attribute__((address_space(3)))

static __device__ __forceinline__ void wait_async_all() {
#if USE_ASYNC_COPY
#if HAVE_WAIT_ASYNC_BUILTIN
  __builtin_amdgcn_s_wait_asynccnt(0);
#else
  asm volatile("s_wait_asynccnt 0" ::: "memory");
#endif
#endif
}

// ---------------------------------------------------------------------------
// Fragment loaders matching the CDNA5 WMMA VGPR layouts (CDNA5 ISA)
//
// 16-bit A (16x32, MxK): lane L holds row M=L&15.
//   this code: lanes 0-15 K 0..15, lanes 16-31 K 16..31 (not ISA table order; B same)
// -> per lane: halves 0..7 = mem[p..p+7], halves 8..15 = mem[p+8..p+15]
static __device__ __forceinline__ v16h load_a_frag(const _Float16* p) {
  v8h lo = *(const v8h*)(p);
  v8h hi = *(const v8h*)(p + 8);
  v16h r;
#pragma unroll
  for (int i = 0; i < 8; ++i) { r[i] = lo[i]; r[i + 8] = hi[i]; }
  return r;
}

// 16-bit B (32x16, KxN): lane L holds column N=L&15; this code puts K=0..15 in lanes 0-15,
// K=16..31 in lanes 16-31 as A does (not the ISA table order), packed 2 per VGPR. With element
// [k][n] = buf[n][k] (row-major along k) this is 16 contiguous halves.
static __device__ __forceinline__ v16h load_b_frag(const _Float16* p) {
  return *(const v16h*)(p);
}

static __device__ __forceinline__ v8f wmma_f16(v16h a, v16h b, v8f c) {
  return __builtin_amdgcn_wmma_f32_16x16x32_f16(
      /*neg_a=*/false, a, /*neg_b=*/false, b,
      /*c_mod=*/(short)0, c, /*reuse_a=*/false, /*reuse_b=*/false);
}

// ---------------------------------------------------------------------------
// f32 -> f16 elementwise convert
__global__ void cvt_f32_f16(const float* __restrict__ src,
                            _Float16* __restrict__ dst, int n) {
  for (int i = blockIdx.x * blockDim.x + threadIdx.x; i < n;
       i += gridDim.x * blockDim.x)
    dst[i] = (_Float16)src[i];
}

// ---------------------------------------------------------------------------
// C[m][n] = sum_k A[m][k] * W[n][k] + bias[n]
// A: [M,K] f16 row-major. W: [N,K] f16 row-major.
// Block = 4 waves; block tile 128x64; wave tile 32x64 (8 WMMA per k-step).
// The shared 64x32 W tile is staged per k-step into double-buffered LDS via
// the async-to-LDS path (ASYNCcnt) when available.
// MODE 0: store f16 row-major [M,N]
// MODE 1: store f16 "V transposed" layout [B,H,DHEAD,S]
// MODE 2: store f32 row-major [M,N] (final output)
template <int MODE>
__global__ __launch_bounds__(128) void gemm_nt_wmma(
    const _Float16* __restrict__ A, const _Float16* __restrict__ W,
    const float* __restrict__ bias, void* __restrict__ out,
    int M, int N, int K) {
  __shared__ __align__(16) _Float16 wtile[2][64 * 32];

  const int tid  = threadIdx.x;
  const int lane = tid & 31;
  const int wave = tid >> 5;
  const int row0 = blockIdx.x * 128 + wave * 32;
  const int n0   = blockIdx.y * 64;
  const int mr   = lane & 15;
  const int hi16 = lane >> 4;

  const _Float16* arow0 = A + (size_t)(row0 + mr) * K + hi16 * 16;
  const _Float16* arow1 = arow0 + (size_t)16 * K;

  v8f acc0[4] = {}, acc1[4] = {};

  // Stage one 64(cols) x 32(k) f16 W tile: 256 16-byte chunks, 2 per thread.
  auto stage = [&](int buf, int kk) {
#pragma unroll
    for (int c = 0; c < 2; ++c) {
      const int ch   = tid * 2 + c;  // 0..255
      const int col  = ch >> 2;      // 0..63
      const int part = ch & 3;       // 0..3 (8 halves each)
      const _Float16* g = W + (size_t)(n0 + col) * K + kk + part * 8;
      _Float16* l = &wtile[buf][col * 32 + part * 8];
#if USE_ASYNC_COPY
      __builtin_amdgcn_global_load_async_to_lds_b128((v4i*)g, (AS3 v4i*)l, 0, 0);
#else
      *(v8h*)l = *(const v8h*)g;
#endif
    }
  };

  stage(0, 0);
  int cur = 0;
  for (int kk = 0; kk < K; kk += 32) {
    wait_async_all();
    __syncthreads();  // current buffer visible to all waves
    if (kk + 32 < K) stage(cur ^ 1, kk + 32);  // prefetch next tile (other buf)

    v16h af0 = load_a_frag(arow0 + kk);
    v16h af1 = load_a_frag(arow1 + kk);
    __builtin_prefetch(arow0 + kk + 128, 0, 0);  // global_prefetch_b8
#pragma unroll
    for (int t = 0; t < 4; ++t) {
      const v16h bf =
          *(const v16h*)&wtile[cur][(t * 16 + mr) * 32 + hi16 * 16];
      acc0[t] = wmma_f16(af0, bf, acc0[t]);
      acc1[t] = wmma_f16(af1, bf, acc1[t]);
    }
    cur ^= 1;
  }

  // C/D layout: lane holds col n = lane&15; VGPR r = row r (lanes 0-15)
  // or row r+8 (lanes 16-31).
  const int mb = hi16 * 8;
#pragma unroll
  for (int u = 0; u < 2; ++u) {
#pragma unroll
    for (int t = 0; t < 4; ++t) {
      const int n = n0 + t * 16 + mr;
      const float bv = bias[n];
#pragma unroll
      for (int r = 0; r < 8; ++r) {
        const int m = row0 + u * 16 + mb + r;
        const float v = (u ? acc1[t][r] : acc0[t][r]) + bv;
        if (MODE == 0) {
          ((_Float16*)out)[(size_t)m * N + n] = (_Float16)v;
        } else if (MODE == 1) {
          const int b = m >> 11, s = m & (SEQ - 1);   // S = 2048
          const int h = n >> 6, d = n & (DHEAD - 1);  // d_head = 64
          ((_Float16*)out)[(((size_t)(b * NHEAD + h) * DHEAD + d) << 11) | s] =
              (_Float16)v;
        } else {
          ((float*)out)[(size_t)m * N + n] = v;
        }
      }
    }
  }
}

// ---------------------------------------------------------------------------
// Flash attention: one wave handles 16 query rows of one (b,h) head.
// Q,K: f16 [B,S,D_MODEL] (head at column offset h*64). Vt: f16 [B,H,64,S].
// O: f16 [B,S,D_MODEL]. Row sums are accumulated with an extra WMMA against
// an all-ones B operand (the "ones column" trick) instead of lane shuffles.
__global__ __launch_bounds__(128) void flash_attn_wmma(
    const _Float16* __restrict__ Q, const _Float16* __restrict__ Kc,
    const _Float16* __restrict__ Vt, _Float16* __restrict__ O) {
  __shared__ __align__(16) _Float16 pbuf[4 * 16 * 32];

  const int lane = threadIdx.x & 31;
  const int wave = threadIdx.x >> 5;
  const int qt   = blockIdx.x * 4 + wave;  // 0..127 (query tile)
  const int bh   = blockIdx.y;             // 0..63
  const int b    = bh >> 4;
  const int h    = bh & (NHEAD - 1);
  const int row0 = qt * 16;
  const int mr   = lane & 15;
  const int hi16 = lane >> 4;

  const _Float16* Qb = Q + ((size_t)b * SEQ + row0) * D_MODEL + h * DHEAD;
  const _Float16* Kb = Kc + (size_t)b * SEQ * D_MODEL + h * DHEAD;
  const _Float16* Vb = Vt + (size_t)bh * DHEAD * SEQ;

  // Q fragments for d=0..31 and d=32..63 (loaded once, reused all k-steps)
  const _Float16* qrow = Qb + (size_t)mr * D_MODEL + hi16 * 16;
  const v16h qf0 = load_a_frag(qrow);
  const v16h qf1 = load_a_frag(qrow + 32);

  v16h ones;
#pragma unroll
  for (int i = 0; i < 16; ++i) ones[i] = (_Float16)1.0f;

  float mrow[8];
  v8f oacc[4] = {};
  v8f lacc = {};  // running softmax denominators (every lane: its row's sum)
#pragma unroll
  for (int r = 0; r < 8; ++r) mrow[r] = -1e30f;

  _Float16* pl = pbuf + wave * (16 * 32);
  const float scale = 0.125f;  // 1/sqrt(64)

  for (int kb = 0; kb < SEQ; kb += 32) {
    // --- scores: two 16x16 tiles over 32 key columns, reduce over d=64 ---
    v8f s0 = {}, s1 = {};
    const _Float16* kp0 = Kb + (size_t)(kb + mr) * D_MODEL + hi16 * 16;
    const _Float16* kp1 = Kb + (size_t)(kb + 16 + mr) * D_MODEL + hi16 * 16;
    s0 = wmma_f16(qf0, load_b_frag(kp0), s0);
    s0 = wmma_f16(qf1, load_b_frag(kp0 + 32), s0);
    s1 = wmma_f16(qf0, load_b_frag(kp1), s1);
    s1 = wmma_f16(qf1, load_b_frag(kp1 + 32), s1);

    // --- online softmax over these 32 columns ---
    float tm[8];
#pragma unroll
    for (int r = 0; r < 8; ++r) {
      s0[r] *= scale;
      s1[r] *= scale;
      tm[r] = fmaxf(s0[r], s1[r]);
    }
    // running-max row reduction across the 16 lanes holding each row
    // (xor 1,2,4,8 stays within each 16-lane half of the wave32)
#pragma unroll
    for (int r = 0; r < 8; ++r) {
      tm[r] = fmaxf(tm[r], __shfl_xor(tm[r], 1, 32));
      tm[r] = fmaxf(tm[r], __shfl_xor(tm[r], 2, 32));
      tm[r] = fmaxf(tm[r], __shfl_xor(tm[r], 4, 32));
      tm[r] = fmaxf(tm[r], __shfl_xor(tm[r], 8, 32));
    }
    float alpha[8];
#pragma unroll
    for (int r = 0; r < 8; ++r) {
      const float mn = fmaxf(mrow[r], tm[r]);
      alpha[r] = __expf(mrow[r] - mn);
      mrow[r] = mn;
    }
#pragma unroll
    for (int r = 0; r < 8; ++r) {
      s0[r] = __expf(s0[r] - mrow[r]);
      s1[r] = __expf(s1[r] - mrow[r]);
    }
    // rescale running accumulators (O and l) by alpha
#pragma unroll
    for (int r = 0; r < 8; ++r) {
      lacc[r] *= alpha[r];
#pragma unroll
      for (int t = 0; t < 4; ++t) oacc[t][r] *= alpha[r];
    }

    // --- relayout P (C-layout) -> A-fragment via LDS ---
    // C layout: lane holds col, VGPR holds row. A layout: lane holds row,
    // VGPR holds K. Stage through this wave's private LDS slice; LDS ops from
    // one wave execute in order (compiler reorder blocked by asm clobber).
    const int rowadd = hi16 * 8;
#pragma unroll
    for (int r = 0; r < 8; ++r) {
      pl[(rowadd + r) * 32 + mr] = (_Float16)s0[r];
      pl[(rowadd + r) * 32 + 16 + mr] = (_Float16)s1[r];
    }
    asm volatile("" ::: "memory");
    const v16h pf = load_a_frag(pl + mr * 32 + hi16 * 16);
    asm volatile("" ::: "memory");

    // --- l += P @ ones ; O += P @ V ---
    lacc = wmma_f16(pf, ones, lacc);
#pragma unroll
    for (int t = 0; t < 4; ++t) {
      const _Float16* vp = Vb + (size_t)(t * 16 + mr) * SEQ + kb + hi16 * 16;
      oacc[t] = wmma_f16(pf, load_b_frag(vp), oacc[t]);
    }
  }

  // --- epilogue: divide by row sums, store f16 [B,S,D_MODEL] ---
  _Float16* Ob = O + ((size_t)b * SEQ + row0) * D_MODEL + h * DHEAD;
  const int rowadd = hi16 * 8;
#pragma unroll
  for (int r = 0; r < 8; ++r) {
    const float inv = 1.0f / lacc[r];
#pragma unroll
    for (int t = 0; t < 4; ++t) {
      Ob[(size_t)(rowadd + r) * D_MODEL + t * 16 + mr] =
          (_Float16)(oacc[t][r] * inv);
    }
  }
}

// ---------------------------------------------------------------------------
extern "C" void kernel_launch(void* const* d_in, const int* in_sizes, int n_in,
                              void* d_out, int out_size, void* d_ws,
                              size_t ws_size, hipStream_t stream) {
  const float* x_q  = (const float*)d_in[0];
  const float* x_kv = (const float*)d_in[1];
  const float* Wq   = (const float*)d_in[2];
  const float* bq   = (const float*)d_in[3];
  const float* Wk   = (const float*)d_in[4];
  const float* bk   = (const float*)d_in[5];
  const float* Wv   = (const float*)d_in[6];
  const float* bv   = (const float*)d_in[7];
  const float* Wo   = (const float*)d_in[8];
  const float* bo   = (const float*)d_in[9];

  char* ws = (char*)d_ws;
  const size_t MB = 1ull << 20;
  _Float16* xq16  = (_Float16*)(ws + 0 * MB);   // 16 MB (reused as attn out)
  _Float16* xkv16 = (_Float16*)(ws + 16 * MB);  // 16 MB
  _Float16* q16   = (_Float16*)(ws + 32 * MB);  // 16 MB
  _Float16* k16   = (_Float16*)(ws + 48 * MB);  // 16 MB
  _Float16* vt16  = (_Float16*)(ws + 64 * MB);  // 16 MB
  _Float16* wq16  = (_Float16*)(ws + 80 * MB);  // 2 MB each
  _Float16* wk16  = (_Float16*)(ws + 82 * MB);
  _Float16* wv16  = (_Float16*)(ws + 84 * MB);
  _Float16* wo16  = (_Float16*)(ws + 86 * MB);
  _Float16* attn16 = xq16;  // x_q f16 copy is dead after the Q projection

  const int NX = BATCH * SEQ * D_MODEL;  // 8388608
  const int NW = D_MODEL * D_MODEL;      // 1048576
  const int M  = BATCH * SEQ;            // 8192

  cvt_f32_f16<<<4096, 256, 0, stream>>>(x_q, xq16, NX);
  cvt_f32_f16<<<4096, 256, 0, stream>>>(x_kv, xkv16, NX);
  cvt_f32_f16<<<1024, 256, 0, stream>>>(Wq, wq16, NW);
  cvt_f32_f16<<<1024, 256, 0, stream>>>(Wk, wk16, NW);
  cvt_f32_f16<<<1024, 256, 0, stream>>>(Wv, wv16, NW);
  cvt_f32_f16<<<1024, 256, 0, stream>>>(Wo, wo16, NW);

  dim3 gg(M / 128, D_MODEL / 64);  // (64, 16), 128 threads = 4 waves
  gemm_nt_wmma<0><<<gg, 128, 0, stream>>>(xq16, wq16, bq, q16, M, D_MODEL, D_MODEL);
  gemm_nt_wmma<0><<<gg, 128, 0, stream>>>(xkv16, wk16, bk, k16, M, D_MODEL, D_MODEL);
  gemm_nt_wmma<1><<<gg, 128, 0, stream>>>(xkv16, wv16, bv, vt16, M, D_MODEL, D_MODEL);

  flash_attn_wmma<<<dim3(SEQ / 64, BATCH * NHEAD), 128, 0, stream>>>(
      q16, k16, vt16, attn16);

  gemm_nt_wmma<2><<<gg, 128, 0, stream>>>(attn16, wo16, bo, d_out, M, D_MODEL,
                                          D_MODEL);
}
